// CausalSelfAttention_32143535243630
// MI455X (gfx1250) — matrix operand order repaired and checked by reading, not compiled, not run on hardware
//
#include <hip/hip_runtime.h>

typedef unsigned short u16;
typedef unsigned int u32;

typedef __attribute__((ext_vector_type(16))) __bf16 v16bf;
typedef __attribute__((ext_vector_type(8)))  float  v8f;
typedef __attribute__((ext_vector_type(4)))  int    v4i;

union FragB16 { u32 u[8]; v16bf v; };
union FragF32 { float f[8]; v8f v; };

// ---------------------------------------------------------------------------
// problem constants
// ---------------------------------------------------------------------------
#define BATCH 2
#define TLEN  2048
#define CDIM  1024
#define HEADS 16
#define HDIM  64
#define C3    (3 * CDIM)
#define MROWS (BATCH * TLEN)          // 4096

// ---------------------------------------------------------------------------
// CDNA5 async global->LDS copy (ASYNCcnt-tracked), with sync fallback
// ---------------------------------------------------------------------------
#if defined(__has_builtin)
#  if __has_builtin(__builtin_amdgcn_global_load_async_to_lds_b128)
#    define HAVE_ASYNC 1
#  endif
#endif
#ifndef HAVE_ASYNC
#  define HAVE_ASYNC 0
#endif

__device__ __forceinline__ void copy16_to_lds(const u16* g, u16* l) {
#if HAVE_ASYNC
    __builtin_amdgcn_global_load_async_to_lds_b128(
        (__attribute__((address_space(1))) v4i*)g,
        (__attribute__((address_space(3))) v4i*)l, 0, 0);
#else
    *(uint4*)l = *(const uint4*)g;
#endif
}

__device__ __forceinline__ void wait_async() {
#if HAVE_ASYNC
#  if __has_builtin(__builtin_amdgcn_s_wait_asynccnt)
    __builtin_amdgcn_s_wait_asynccnt(0);
#  else
    asm volatile("s_wait_asynccnt 0x0" ::: "memory");
#  endif
#endif
}

// round-to-nearest-even f32 -> bf16 bits
__device__ __forceinline__ u16 f2bf(float x) {
    u32 u = __float_as_uint(x);
    u32 r = u + 0x7FFFu + ((u >> 16) & 1u);
    return (u16)(r >> 16);
}

// A-matrix (16x32 bf16) per-lane K index for VGPR j, lane-half h
__device__ __forceinline__ int kA_of(int j, int h) {
    return (j < 4 ? 2 * j : 16 + 2 * (j - 4)) + 8 * h;
}
// B-matrix (32x16 bf16): lane = N, VGPR j holds same K pair as A (kA_of)
__device__ __forceinline__ int kB_of(int j, int h) {
    return (j < 4 ? 2 * j : 16 + 2 * (j - 4)) + 8 * h;
}

__device__ __forceinline__ v8f wmma_bf16(const FragB16& a, const FragB16& b, v8f c) {
    return __builtin_amdgcn_wmma_f32_16x16x32_bf16(
        false, a.v, false, b.v, (short)0, c, false, false);
}

// ---------------------------------------------------------------------------
// elementwise f32 -> bf16
// ---------------------------------------------------------------------------
__global__ __launch_bounds__(256) void k_f32_to_bf16(
    const float* __restrict__ src, u16* __restrict__ dst, int n) {
    int i = blockIdx.x * 256 + threadIdx.x;
    if (i < n) dst[i] = f2bf(src[i]);
}

// Wt[n*K + k] = bf16(W[k*N + n])
__global__ __launch_bounds__(256) void k_transpose_bf16(
    const float* __restrict__ W, u16* __restrict__ Wt, int K, int N) {
    long long i = (long long)blockIdx.x * 256 + threadIdx.x;
    if (i < (long long)N * K) {
        int k = (int)(i % K);
        int n = (int)(i / K);
        Wt[i] = f2bf(W[(size_t)k * N + n]);
    }
}

// ---------------------------------------------------------------------------
// GEMM: C[M,N] = A[M,K](bf16) * Bt[N,K](bf16)^T + bias[N]
// 128x128 block, 8 waves (wave tile 64x32), K staged 64 per iteration,
// double-buffered LDS fed by async global->LDS copies.
// ---------------------------------------------------------------------------
template <bool OUT_BF16>
__global__ __launch_bounds__(256) void k_gemm_wmma(
    const u16* __restrict__ A, const u16* __restrict__ Bt,
    const float* __restrict__ bias, void* __restrict__ Cout,
    int M, int N, int K) {
    const int LDT = 72;   // padded row (elements); 144B stride, conflict-free frags
    __shared__ u16 As[2][128 * LDT];
    __shared__ u16 Bs[2][128 * LDT];

    const int tid  = threadIdx.x;
    const int wave = tid >> 5;
    const int lane = tid & 31;
    const int hl   = lane >> 4;
    const int ln   = lane & 15;
    const int wm   = wave >> 2;    // 0..1 -> 64-row strip
    const int wn   = wave & 3;     // 0..3 -> 32-col strip
    const int m0   = blockIdx.y * 128;
    const int n0   = blockIdx.x * 128;

    FragF32 acc[4][2];
#pragma unroll
    for (int mt = 0; mt < 4; ++mt)
#pragma unroll
        for (int nt = 0; nt < 2; ++nt)
#pragma unroll
            for (int r = 0; r < 8; ++r) acc[mt][nt].f[r] = 0.0f;

    const int ldRow = tid >> 1;          // 0..127
    const int ldK   = (tid & 1) * 16;    // 0 or 16
    const u16* gA = A  + (size_t)(m0 + ldRow) * K + ldK;
    const u16* gB = Bt + (size_t)(n0 + ldRow) * K + ldK;

    auto stage = [&](int k0, int b) {
        u16* la = &As[b][ldRow * LDT + ldK];
        u16* lb = &Bs[b][ldRow * LDT + ldK];
        copy16_to_lds(gA + k0,      la);
        copy16_to_lds(gA + k0 + 32, la + 32);
        copy16_to_lds(gB + k0,      lb);
        copy16_to_lds(gB + k0 + 32, lb + 32);
    };

    stage(0, 0);
    wait_async();
    __syncthreads();

    int buf = 0;
    for (int k0 = 0; k0 < K; k0 += 64) {
        if (k0 + 64 < K) stage(k0 + 64, buf ^ 1);   // overlaps with compute below
#pragma unroll
        for (int ks = 0; ks < 64; ks += 32) {
            FragB16 af[4], bf[2];
#pragma unroll
            for (int mt = 0; mt < 4; ++mt) {
                int row = wm * 64 + mt * 16 + ln;
#pragma unroll
                for (int j = 0; j < 8; ++j)
                    af[mt].u[j] = *(const u32*)&As[buf][row * LDT + ks + kA_of(j, hl)];
            }
#pragma unroll
            for (int nt = 0; nt < 2; ++nt) {
                int col = wn * 32 + nt * 16 + ln;
#pragma unroll
                for (int j = 0; j < 8; ++j)
                    bf[nt].u[j] = *(const u32*)&Bs[buf][col * LDT + ks + kB_of(j, hl)];
            }
#pragma unroll
            for (int mt = 0; mt < 4; ++mt)
#pragma unroll
                for (int nt = 0; nt < 2; ++nt)
                    acc[mt][nt].v = wmma_bf16(af[mt], bf[nt], acc[mt][nt].v);
        }
        wait_async();        // next-stage copies finished (overlapped with WMMAs)
        __syncthreads();
        buf ^= 1;
    }

    // epilogue: C layout -> row = r + 8*hl, col = ln per 16x16 tile
#pragma unroll
    for (int nt = 0; nt < 2; ++nt) {
        int col = n0 + wn * 32 + nt * 16 + ln;
        float bv = bias[col];
#pragma unroll
        for (int mt = 0; mt < 4; ++mt) {
#pragma unroll
            for (int r = 0; r < 8; ++r) {
                int row = m0 + wm * 64 + mt * 16 + r + 8 * hl;
                float v = acc[mt][nt].f[r] + bv;
                if (OUT_BF16)
                    ((u16*)Cout)[(size_t)row * N + col] = f2bf(v);
                else
                    ((float*)Cout)[(size_t)row * N + col] = v;
            }
        }
    }
}

// ---------------------------------------------------------------------------
// Flash attention: one WG per (b, h, 128 q rows); 8 waves, 16 q rows each.
// ---------------------------------------------------------------------------
__global__ __launch_bounds__(256) void k_attention(
    const u16* __restrict__ qkv, u16* __restrict__ Y) {
    __shared__ u16 Ks[32 * 72];       // K tile row-major [key][d], stride 72
    __shared__ u16 Vt[64 * 34];       // V tile transposed [d][key], stride 34
    __shared__ u16 Pw[8][16 * 34];    // per-wave P staging

    const int tid  = threadIdx.x;
    const int wave = tid >> 5;
    const int lane = tid & 31;
    const int hl   = lane >> 4;
    const int ln   = lane & 15;
    const int qb   = blockIdx.x;
    const int bh   = blockIdx.y;
    const int b    = bh >> 4;
    const int h    = bh & 15;
    const int qbase = qb * 128 + wave * 16;
    // scalar copy of the (wave-uniform) q-tile base for uniform branches
    const int qbase_s = __builtin_amdgcn_readfirstlane(qbase);

    // Q fragments (16x64): two 16x32 A-frags straight from global
    FragB16 aq[2];
#pragma unroll
    for (int f = 0; f < 2; ++f)
#pragma unroll
        for (int j = 0; j < 8; ++j) {
            int k = f * 32 + kA_of(j, hl);
            size_t idx = (size_t)(b * TLEN + qbase + ln) * C3 + h * HDIM + k;
            aq[f].u[j] = *(const u32*)(qkv + idx);
        }

    // constant ones B-fragment: rowsum(P) = P @ ones via one WMMA
    FragB16 bone;
#pragma unroll
    for (int j = 0; j < 8; ++j) bone.u[j] = 0x3F803F80u;   // bf16 1.0 pairs
    const v8f vzero = {};

    float m_r[8], l_r[8];
    FragF32 o[4];
#pragma unroll
    for (int r = 0; r < 8; ++r) { m_r[r] = -INFINITY; l_r[r] = 0.0f; }
#pragma unroll
    for (int dt = 0; dt < 4; ++dt)
#pragma unroll
        for (int r = 0; r < 8; ++r) o[dt].f[r] = 0.0f;

    const int ldN = tid >> 3;          // key slot 0..31
    const int ldD = (tid & 7) * 8;     // d offset 0..56

    const int ntiles = qb * 4 + 4;
    for (int jk = 0; jk < ntiles; ++jk) {
        // stage K tile (async copy) and V tile (manual transpose)
        {
            int tok = jk * 32 + ldN;
            size_t gk = (size_t)(b * TLEN + tok) * C3 + CDIM + h * HDIM + ldD;
            size_t gv = (size_t)(b * TLEN + tok) * C3 + 2 * CDIM + h * HDIM + ldD;
            copy16_to_lds(qkv + gk, &Ks[ldN * 72 + ldD]);
            uint4 vd = *(const uint4*)(qkv + gv);
            const u16* vp = (const u16*)&vd;
#pragma unroll
            for (int i = 0; i < 8; ++i) Vt[(ldD + i) * 34 + ldN] = vp[i];
            if (jk + 1 < ntiles)
                __builtin_prefetch(qkv + gv + (size_t)32 * C3, 0, 1);
        }
        wait_async();
        __syncthreads();

        if (jk * 32 <= qbase_s) {                     // scalar, wave-uniform
            // S = q @ k^T : two 16x16 key sub-tiles, contraction over D=64
            FragF32 s[2];
#pragma unroll
            for (int nt = 0; nt < 2; ++nt) {
#pragma unroll
                for (int r = 0; r < 8; ++r) s[nt].f[r] = 0.0f;
                FragB16 bk;
#pragma unroll
                for (int f = 0; f < 2; ++f) {
#pragma unroll
                    for (int j = 0; j < 8; ++j) {
                        int d = f * 32 + kB_of(j, hl);
                        int n = nt * 16 + ln;
                        bk.u[j] = *(const u32*)&Ks[n * 72 + d];
                    }
                    s[nt].v = wmma_bf16(aq[f], bk, s[nt].v);
                }
            }
            // row max (raw domain); causal mask only on the diagonal tile
            float rmax[8];
            if (jk * 32 + 31 > qbase_s) {             // masked (diagonal) tile
#pragma unroll
                for (int r = 0; r < 8; ++r) {
                    int mrow = qbase + r + 8 * hl;
                    float mx = -INFINITY;
#pragma unroll
                    for (int nt = 0; nt < 2; ++nt) {
                        int ncol = jk * 32 + nt * 16 + ln;
                        float v = (ncol > mrow) ? -INFINITY : s[nt].f[r];
                        s[nt].f[r] = v;
                        mx = fmaxf(mx, v);
                    }
#pragma unroll
                    for (int off = 1; off < 16; off <<= 1)
                        mx = fmaxf(mx, __shfl_xor(mx, off, 32));
                    rmax[r] = mx * 8.0f;
                }
            } else {                                  // interior tile: no mask
#pragma unroll
                for (int r = 0; r < 8; ++r) {
                    float mx = fmaxf(s[0].f[r], s[1].f[r]);
#pragma unroll
                    for (int off = 1; off < 16; off <<= 1)
                        mx = fmaxf(mx, __shfl_xor(mx, off, 32));
                    rmax[r] = mx * 8.0f;
                }
            }
            // online softmax rescale; exp argument via single fma (8*s - m)
#pragma unroll
            for (int r = 0; r < 8; ++r) {
                float mnew = fmaxf(m_r[r], rmax[r]);
                float corr = __expf(m_r[r] - mnew);
                m_r[r] = mnew;
                l_r[r] *= corr;
#pragma unroll
                for (int dt = 0; dt < 4; ++dt) o[dt].f[r] *= corr;
#pragma unroll
                for (int nt = 0; nt < 2; ++nt)
                    s[nt].f[r] = __expf(__builtin_fmaf(s[nt].f[r], 8.0f, -mnew));
            }
            // P: C layout -> LDS -> A layout (same-wave DS is in-order)
            u16* pp = Pw[wave];
#pragma unroll
            for (int r = 0; r < 8; ++r)
#pragma unroll
                for (int nt = 0; nt < 2; ++nt)
                    pp[(r + 8 * hl) * 34 + nt * 16 + ln] = f2bf(s[nt].f[r]);
            asm volatile("" ::: "memory");
            FragB16 ap;
#pragma unroll
            for (int j = 0; j < 8; ++j)
                ap.u[j] = *(const u32*)&pp[ln * 34 + kA_of(j, hl)];
            // l += rowsum(P) with one WMMA against the ones matrix
            {
                FragF32 ls;
                ls.v = wmma_bf16(ap, bone, vzero);
#pragma unroll
                for (int r = 0; r < 8; ++r) l_r[r] += ls.f[r];
            }
            // O += P @ V
#pragma unroll
            for (int dt = 0; dt < 4; ++dt) {
                FragB16 bv;
#pragma unroll
                for (int j = 0; j < 8; ++j) {
                    int n = kB_of(j, hl);
                    int d = dt * 16 + ln;
                    bv.u[j] = *(const u32*)&Vt[d * 34 + n];
                }
                o[dt].v = wmma_bf16(ap, bv, o[dt].v);
            }
        }
        __syncthreads();
    }

    // normalize + store bf16 (head-merged [B*T, C])
#pragma unroll
    for (int dt = 0; dt < 4; ++dt)
#pragma unroll
        for (int r = 0; r < 8; ++r) {
            int row = qbase + r + 8 * hl;
            int col = dt * 16 + ln;
            float v = o[dt].f[r] / l_r[r];
            Y[(size_t)(b * TLEN + row) * CDIM + h * HDIM + col] = f2bf(v);
        }
}

// ---------------------------------------------------------------------------
// launch
// ---------------------------------------------------------------------------
extern "C" void kernel_launch(void* const* d_in, const int* in_sizes, int n_in,
                              void* d_out, int out_size, void* d_ws, size_t ws_size,
                              hipStream_t stream) {
    const float* x  = (const float*)d_in[0];
    const float* Wa = (const float*)d_in[1];
    const float* ba = (const float*)d_in[2];
    const float* Wp = (const float*)d_in[3];
    const float* bp = (const float*)d_in[4];
    float* out = (float*)d_out;

    char* ws = (char*)d_ws;
    u16* Xbf  = (u16*)ws;                                  ws += (size_t)MROWS * CDIM * 2;
    u16* WaT  = (u16*)ws;                                  ws += (size_t)C3 * CDIM * 2;
    u16* WpT  = (u16*)ws;                                  ws += (size_t)CDIM * CDIM * 2;
    u16* QKV  = (u16*)ws;                                  ws += (size_t)MROWS * C3 * 2;
    u16* Ybf  = (u16*)ws;

    k_f32_to_bf16<<<(MROWS * CDIM) / 256, 256, 0, stream>>>(x, Xbf, MROWS * CDIM);
    k_transpose_bf16<<<(C3 * CDIM) / 256, 256, 0, stream>>>(Wa, WaT, CDIM, C3);
    k_transpose_bf16<<<(CDIM * CDIM) / 256, 256, 0, stream>>>(Wp, WpT, CDIM, CDIM);

    k_gemm_wmma<true><<<dim3(C3 / 128, MROWS / 128), 256, 0, stream>>>(
        Xbf, WaT, ba, (void*)QKV, MROWS, C3, CDIM);

    k_attention<<<dim3(TLEN / 128, BATCH * HEADS), 256, 0, stream>>>(QKV, Ybf);

    k_gemm_wmma<false><<<dim3(CDIM / 128, MROWS / 128), 256, 0, stream>>>(
        Ybf, WpT, bp, (void*)out, MROWS, CDIM, CDIM);
}
